// CurrentLIFNetwork_IO_45938970198743
// MI455X (gfx1250) — matrix operand order repaired and checked by reading, not compiled, not run on hardware
//
#include <hip/hip_runtime.h>

// ---------------------------------------------------------------------------
// LIF spiking network, B=16, T=500, NIN=1024, N=4096, S=2 (exc/inh).
// Strategy: batch M=16 == one WMMA tile row. Per timestep:
//   drive_s = spk_masked_s[16xN] @ Wsigned[NxN]   (f16 WMMA, f32 acc)
//   Iff     = spk_in[16xNIN]     @ W_FF[NINxN]    (f16 WMMA, f32 acc)
// Wsigned / W_FF are pre-packed (once) into the per-lane B-operand fragment
// layout of v_wmma_f32_16x16x32_f16 so the inner loop is pure b128 loads +
// WMMA. Recurrence requires a global barrier per step -> 500 step-kernel
// launches on `stream` (graph-capture safe). Spike buffers double-buffered.
// W (32 MB f16) + W_FF (8 MB f16) stay resident in the 192 MB L2; prefetch
// at WGP scope (locality=3) pulls the streaming B tiles into WGP$ ahead of
// the WMMA consumer.
// ---------------------------------------------------------------------------

typedef __attribute__((ext_vector_type(16))) _Float16 v16h;
typedef __attribute__((ext_vector_type(8)))  _Float16 v8h;
typedef __attribute__((ext_vector_type(8)))  float    v8f;

#define NB   16
#define NT   500
#define NNIN 1024
#define NN   4096
#define CT_TILES 256          /* NN/16 column tiles   */
#define KT_REC   128          /* NN/32 K tiles        */
#define KT_FF    32           /* NIN/32 K tiles       */

// Workspace layout (bytes). Total ~57.4 MB.
#define WS_WPACK   ((size_t)0)                       // 256*128*512 halfs = 32 MB
#define WS_WFFPACK ((size_t)33554432)                // 256*32*512 halfs  =  8 MB
#define WS_INF16   ((size_t)41943040)                // 16*500*1024 halfs = 15.6 MB
#define WS_STATE   ((size_t)58327040)                // 5 x 16*4096 f32   = 1.25 MB
#define WS_SPK     ((size_t)59637760)                // 4 x 16*4096 f16   = 0.5 MB

union V16U { v16h v; v8h h[2]; };

// A-operand fragment (16x32 f16, row-major source): lane holds row M=lane%16,
// K halves {base..base+15} in v0..7,
// base = (lane<16 ? 0 : 16).  Two 16B loads, second at +8 halfs.
__device__ __forceinline__ v16h ldfragA(const _Float16* p) {
  V16U u;
  u.h[0] = *(const v8h*)(p);
  u.h[1] = *(const v8h*)(p + 8);
  return u.v;
}
// B-operand fragment from pre-packed buffer: lane's 16 halfs are contiguous.
__device__ __forceinline__ v16h ldfragB(const _Float16* p) {
  V16U u;
  u.h[0] = *(const v8h*)(p);
  u.h[1] = *(const v8h*)(p + 8);
  return u.v;
}

// ---------------------------------------------------------------------------
// Prep: zero state + initial spike buffers.
// ---------------------------------------------------------------------------
__global__ void lif_zero(float* I0, float* I1, float* IF, float* V, float* RF,
                         _Float16* se0, _Float16* si0,
                         _Float16* se1, _Float16* si1) {
  int i = blockIdx.x * blockDim.x + threadIdx.x;
  if (i < NB * NN) {
    I0[i] = 0.f; I1[i] = 0.f; IF[i] = 0.f; V[i] = 0.f; RF[i] = 0.f;
    se0[i] = (_Float16)0.f; si0[i] = (_Float16)0.f;
    se1[i] = (_Float16)0.f; si1[i] = (_Float16)0.f;
  }
}

// Pack signed recurrent weights into WMMA-B fragment layout.
// Block (ct,kt) = 32 lanes x 16 halfs; lane l covers column n = ct*16+(l%16),
// K = kt*32 + (l<16?0:16) + j, j=0..15.  Sign: +W for exc (type 0), -W inh.
__global__ void lif_pack_w(const float* __restrict__ W,
                           const int* __restrict__ ctype,
                           _Float16* __restrict__ Wp) {
  int gid  = blockIdx.x * blockDim.x + threadIdx.x;
  int blk  = gid >> 5;
  int lane = gid & 31;
  if (blk >= CT_TILES * KT_REC) return;
  int ct = blk / KT_REC, kt = blk % KT_REC;
  int n     = ct * 16 + (lane & 15);
  int kbase = kt * 32 + ((lane & 16) ? 16 : 0);
  _Float16* dst = Wp + (size_t)blk * 512 + lane * 16;
  #pragma unroll
  for (int j = 0; j < 16; ++j) {
    int k = kbase + j;
    float w = W[(size_t)k * NN + n];
    dst[j] = (_Float16)((ctype[k] != 0) ? -w : w);
  }
}

// Pack feedforward weights (no sign).
__global__ void lif_pack_wff(const float* __restrict__ WFF,
                             _Float16* __restrict__ Wp) {
  int gid  = blockIdx.x * blockDim.x + threadIdx.x;
  int blk  = gid >> 5;
  int lane = gid & 31;
  if (blk >= CT_TILES * KT_FF) return;
  int ct = blk / KT_FF, kt = blk % KT_FF;
  int n     = ct * 16 + (lane & 15);
  int kbase = kt * 32 + ((lane & 16) ? 16 : 0);
  _Float16* dst = Wp + (size_t)blk * 512 + lane * 16;
  #pragma unroll
  for (int j = 0; j < 16; ++j)
    dst[j] = (_Float16)WFF[(size_t)(kbase + j) * NN + n];
}

// Convert input spikes [B,T,NIN] f32 -> f16 (0/1 exact).
__global__ void lif_cvt_in(const float* __restrict__ src,
                           _Float16* __restrict__ dst) {
  int i = blockIdx.x * blockDim.x + threadIdx.x;
  if (i < NB * NT * NNIN) dst[i] = (_Float16)src[i];
}

// ---------------------------------------------------------------------------
// One timestep. Grid: 32 blocks x 256 threads = 256 waves; each wave owns one
// 16-column tile. D-matrix layout: VGPR r, lane l -> batch b = r + 8*(l>=16),
// column n = ct*16 + (l%16).
// ---------------------------------------------------------------------------
__global__ void __launch_bounds__(256)
lif_step(const _Float16* __restrict__ Wp, const _Float16* __restrict__ Wffp,
         const _Float16* __restrict__ inF16, const int* __restrict__ ctype,
         const _Float16* __restrict__ spkE, const _Float16* __restrict__ spkI,
         _Float16* __restrict__ spkEout, _Float16* __restrict__ spkIout,
         float* I0, float* I1, float* IF, float* V, float* RF,
         float* __restrict__ out, int t) {
  const int lane  = threadIdx.x & 31;
  const int wid   = threadIdx.x >> 5;
  const int ct    = blockIdx.x * 8 + wid;           // 0..255 column tile
  const int arow  = lane & 15;                      // A-fragment row (batch)
  const int abase = (lane & 16) ? 16 : 0;           // A-fragment K sub-base

  v8f acc0 = {}; v8f acc1 = {}; v8f accf = {};

  // ---- recurrent: two WMMAs share each B tile (exc / inh masked spikes) ----
  const _Float16* bblk = Wp + (size_t)ct * KT_REC * 512 + lane * 16;
  const _Float16* aE0  = spkE + (size_t)arow * NN + abase;
  const _Float16* aI0  = spkI + (size_t)arow * NN + abase;
  for (int kt = 0; kt < KT_REC; ++kt) {
    // WGP-scope prefetch of the next B tile: pulls L2 -> WGP$ ahead of use.
    __builtin_prefetch(bblk + 512, 0, 3);           // global_prefetch_b8 (WGP)
    v16h bf = ldfragB(bblk);
    v16h ae = ldfragA(aE0 + kt * 32);
    v16h ai = ldfragA(aI0 + kt * 32);
    acc0 = __builtin_amdgcn_wmma_f32_16x16x32_f16(false, ae, false, bf,
                                                  (short)0, acc0, false, false);
    acc1 = __builtin_amdgcn_wmma_f32_16x16x32_f16(false, ai, false, bf,
                                                  (short)0, acc1, false, false);
    bblk += 512;
  }

  // ---- feedforward ----
  const _Float16* fblk  = Wffp + (size_t)ct * KT_FF * 512 + lane * 16;
  const _Float16* inrow = inF16 + ((size_t)arow * NT + t) * NNIN + abase;
  for (int kt = 0; kt < KT_FF; ++kt) {
    __builtin_prefetch(fblk + 512, 0, 3);
    v16h bf = ldfragB(fblk);
    v16h af = ldfragA(inrow + kt * 32);
    accf = __builtin_amdgcn_wmma_f32_16x16x32_f16(false, af, false, bf,
                                                  (short)0, accf, false, false);
    fblk += 512;
  }

  // ---- fused LIF state update ----
  const int   n   = ct * 16 + (lane & 15);
  const int   ty  = ctype[n];
  const float A0  = 0.81873075307798185867f;  // exp(-1/5)  ampa
  const float A1  = 0.90483741803595957316f;  // exp(-1/10) gaba
  const float AF  = 0.81873075307798185867f;  // exp(-1/5)  ff
  const float BET = 0.95122942450071400910f;  // exp(-1/20) membrane
  const float OMB = 1.0f - BET;

  #pragma unroll
  for (int r = 0; r < 8; ++r) {
    const int    b   = r + ((lane & 16) ? 8 : 0);
    const size_t idx = (size_t)b * NN + n;
    float i0  = A0 * I0[idx] + acc0[r];
    float i1  = A1 * I1[idx] + acc1[r];       // inh rows pre-negated in Wp
    float iff = AF * IF[idx] + accf[r];
    float itot = i0 + i1 + iff;
    float ref  = RF[idx];
    bool  active = (ref <= 0.0f);
    float vnew = active ? (BET * V[idx] + OMB * itot) : 0.0f;
    float s    = (active && (vnew - 1.0f > 0.0f)) ? 1.0f : 0.0f;
    float refn = (s > 0.0f) ? 2.0f : fmaxf(ref - 1.0f, 0.0f);
    I0[idx] = i0; I1[idx] = i1; IF[idx] = iff;
    V[idx]  = vnew * (1.0f - s);              // reset on spike
    RF[idx] = refn;
    out[((size_t)b * NT + t) * NN + n] = s;
    spkEout[idx] = (_Float16)((ty == 0) ? s : 0.0f);
    spkIout[idx] = (_Float16)((ty != 0) ? s : 0.0f);
  }
}

// ---------------------------------------------------------------------------
extern "C" void kernel_launch(void* const* d_in, const int* in_sizes, int n_in,
                              void* d_out, int out_size, void* d_ws,
                              size_t ws_size, hipStream_t stream) {
  const float* in_spk = (const float*)d_in[0];   // [B,T,NIN] f32
  const float* W      = (const float*)d_in[1];   // [N,N]     f32
  const float* W_FF   = (const float*)d_in[2];   // [NIN,N]   f32
  const int*   ctype  = (const int*)d_in[3];     // [N]       i32
  float*       out    = (float*)d_out;           // [B,T,N]   f32
  char*        ws     = (char*)d_ws;

  _Float16* Wp   = (_Float16*)(ws + WS_WPACK);
  _Float16* Wffp = (_Float16*)(ws + WS_WFFPACK);
  _Float16* inF  = (_Float16*)(ws + WS_INF16);
  float* I0 = (float*)(ws + WS_STATE);
  float* I1 = I0 + NB * NN;
  float* IF = I1 + NB * NN;
  float* V  = IF + NB * NN;
  float* RF = V  + NB * NN;
  _Float16* spkBase = (_Float16*)(ws + WS_SPK);
  _Float16* se[2] = { spkBase,               spkBase + 2 * NB * NN };
  _Float16* si[2] = { spkBase + NB * NN,     spkBase + 3 * NB * NN };

  lif_pack_w  <<<(CT_TILES * KT_REC * 32) / 256, 256, 0, stream>>>(W, ctype, Wp);
  lif_pack_wff<<<(CT_TILES * KT_FF  * 32) / 256, 256, 0, stream>>>(W_FF, Wffp);
  lif_cvt_in  <<<(NB * NT * NNIN) / 256,        256, 0, stream>>>(in_spk, inF);
  lif_zero    <<<(NB * NN) / 256,               256, 0, stream>>>(
      I0, I1, IF, V, RF, se[0], si[0], se[1], si[1]);

  for (int t = 0; t < NT; ++t) {
    const int cur = t & 1, nxt = cur ^ 1;
    lif_step<<<32, 256, 0, stream>>>(Wp, Wffp, inF, ctype,
                                     se[cur], si[cur], se[nxt], si[nxt],
                                     I0, I1, IF, V, RF, out, t);
  }
}
